// GNNMessageModel_66855460930089
// MI455X (gfx1250) — matrix operand order repaired and checked by reading, not compiled, not run on hardware
//
#include <hip/hip_runtime.h>

typedef __attribute__((ext_vector_type(16))) _Float16 v16h;
typedef __attribute__((ext_vector_type(8)))  float    v8f;

// ---------------------------------------------------------------------------
// Kernel 1: h = relu(z @ enc_W + enc_b)            [N,128]@[128,32] -> [N,32]
// ---------------------------------------------------------------------------
__global__ void enc_kernel(const float* __restrict__ z,
                           const float* __restrict__ W,
                           const float* __restrict__ b,
                           float* __restrict__ h, int n) {
  int gid = blockIdx.x * blockDim.x + threadIdx.x;
  if (gid >= n * 32) return;
  int row = gid >> 5, c = gid & 31;
  const float* zr = z + (size_t)row * 128;
  float s = b[c];
#pragma unroll 8
  for (int k = 0; k < 128; ++k) s = fmaf(zr[k], W[k * 32 + c], s);
  h[gid] = fmaxf(s, 0.0f);
}

// ---------------------------------------------------------------------------
// Kernel 2: messages = h @ msg_W + msg_b. Written three ways:
//  - fp32 copy (for the diagonal fix-up in the aggregation epilogue)
//  - split-f16 hi/lo pair, pre-swizzled into this code's contiguous-k WMMA B order
//    (16x16x32):  lane<16: N=lane,  K = 2v,2v+1  (v = vgpr 0..7)
//                 lane>=16: N=lane-16, K = 16+2v, 16+2v+1
//    slot: j0c=j/32, K=j%32, nt=col/16, e=K&15,
//          lane=(K<16 ? col&15 : 16+(col&15)),
//          o = ((j0c*4 + nt)*32 + lane)*16 + e
// ---------------------------------------------------------------------------
__global__ void msg_pack_kernel(const float* __restrict__ h,
                                const float* __restrict__ W,
                                const float* __restrict__ b,
                                _Float16* __restrict__ packHi,
                                _Float16* __restrict__ packLo,
                                float* __restrict__ msgs, int n) {
  int gid = blockIdx.x * blockDim.x + threadIdx.x;
  if (gid >= n * 64) return;
  int row = gid >> 6, c = gid & 63;
  const float* hr = h + (size_t)row * 32;
  float s = b[c];
#pragma unroll
  for (int k = 0; k < 32; ++k) s = fmaf(hr[k], W[k * 64 + c], s);

  msgs[gid] = s;
  _Float16 hi = (_Float16)s;
  float lo = s - (float)hi;

  int j0c = row >> 5, K = row & 31;
  int lane = (K < 16) ? (c & 15) : (16 + (c & 15));
  int e = K & 15;
  int nt = c >> 4;
  size_t o = ((size_t)(j0c * 4 + nt) * 32 + lane) * 16 + e;
  packHi[o] = hi;
  packLo[o] = (_Float16)lo;
}

// ---------------------------------------------------------------------------
// Kernel 3: aggregated = adj @ messages via v_wmma_f32_16x16x32_f16.
// Block = 32-row strip (two 16-row M-tiles per wave), 8 waves split j.
// Every B hi/lo tile pair feeds 4 independent WMMAs (2 M-tiles x hi/lo), so
// each load wait amortizes over 4 matrix ops.  dists are double-buffered
// (chunk t+1 in flight while chunk t computes).  Adjacency fragment is the
// bare (dist < 1.5) compare; the diagonal term is subtracted exactly in the
// epilogue (agg[i] -= adj_ii * messages[i]).
// A fragment layout used here (contiguous k, same as B pack): lane<16 row M=lane, K 0-15;
// lane>=16 row M=lane-16, K 16-31.
// ---------------------------------------------------------------------------
#define WMMA_F16(C, A, B) \
  C = __builtin_amdgcn_wmma_f32_16x16x32_f16(false, A, false, B, (short)0, C, false, false)

__device__ __forceinline__ v16h adj_frag(const float4& a0, const float4& a1,
                                         const float4& a2, const float4& a3) {
  float dv[16] = {a0.x, a0.y, a0.z, a0.w, a1.x, a1.y, a1.z, a1.w,
                  a2.x, a2.y, a2.z, a2.w, a3.x, a3.y, a3.z, a3.w};
  v16h a;
#pragma unroll
  for (int t = 0; t < 16; ++t)
    a[t] = (dv[t] < 1.5f) ? (_Float16)1.0f : (_Float16)0.0f;
  return a;
}

__global__ void __launch_bounds__(256)
agg_wmma_kernel(const float* __restrict__ dists,
                const _Float16* __restrict__ packHi,
                const _Float16* __restrict__ packLo,
                const float* __restrict__ msgs,
                float* __restrict__ agg, int n) {
  __shared__ float acc[32 * 64];
  const int tid  = threadIdx.x;
  const int lane = tid & 31;
  const int wave = tid >> 5;
  const int i0   = blockIdx.x << 5;     // 32 rows per block

  for (int k = tid; k < 2048; k += 256) acc[k] = 0.0f;
  __syncthreads();

  const int rl    = lane & 15;
  const int kbase = (lane < 16) ? 0 : 16;
  const float* drow0 = dists + (size_t)(i0 + rl) * n + kbase;       // M-tile 0
  const float* drow1 = dists + (size_t)(i0 + 16 + rl) * n + kbase;  // M-tile 1
  const int laneOff = lane * 16;

  v8f c00 = {}, c01 = {}, c02 = {}, c03 = {};   // M-tile 0, N-tiles 0..3
  v8f c10 = {}, c11 = {}, c12 = {}, c13 = {};   // M-tile 1, N-tiles 0..3

  const int jHome = wave * 32;
  const int nIter = n >> 8;             // j-chunks per wave

#define LOAD_D(P, jj, a0, a1, a2, a3)                      \
  { const float* dp = (P) + (jj);                          \
    a0 = *(const float4*)(dp);                             \
    a1 = *(const float4*)(dp + 4);                         \
    a2 = *(const float4*)(dp + 8);                         \
    a3 = *(const float4*)(dp + 12); }

#define LOAD_B(jj, t, H, L)                                            \
  { size_t bo = ((size_t)((jj) >> 5) * 4 + (t)) * 512 + laneOff;       \
    H = *(const v16h*)(packHi + bo);                                   \
    L = *(const v16h*)(packLo + bo); }

  float4 d0, d1, d2, d3, f0, f1, f2, f3;   // dists: current chunk (tiles 0/1)
  float4 e0, e1, e2, e3, g0, g1, g2, g3;   // dists: next chunk in flight

  int j0 = jHome;
  LOAD_D(drow0, j0, d0, d1, d2, d3);
  LOAD_D(drow1, j0, f0, f1, f2, f3);

  for (int it = 0; it < nIter; ++it) {
    const int jn = (it + 1 < nIter) ? (j0 + 256) : jHome;  // wrap = dummy reload

    // next-iteration dists go in flight before any compute of this chunk
    LOAD_D(drow0, jn, e0, e1, e2, e3);
    LOAD_D(drow1, jn, g0, g1, g2, g3);
    __builtin_prefetch(drow0 + jn + 256, 0, 3);
    __builtin_prefetch(drow1 + jn + 256, 0, 3);

    // 0/1 f16 adjacency fragments for the two M-tiles
    v16h a0 = adj_frag(d0, d1, d2, d3);
    v16h a1 = adj_frag(f0, f1, f2, f3);

    // each B pair feeds 4 independent WMMAs
    v16h bh, bl;
    LOAD_B(j0, 0, bh, bl);
    WMMA_F16(c00, a0, bh);  WMMA_F16(c10, a1, bh);
    WMMA_F16(c00, a0, bl);  WMMA_F16(c10, a1, bl);
    LOAD_B(j0, 1, bh, bl);
    WMMA_F16(c01, a0, bh);  WMMA_F16(c11, a1, bh);
    WMMA_F16(c01, a0, bl);  WMMA_F16(c11, a1, bl);
    LOAD_B(j0, 2, bh, bl);
    WMMA_F16(c02, a0, bh);  WMMA_F16(c12, a1, bh);
    WMMA_F16(c02, a0, bl);  WMMA_F16(c12, a1, bl);
    LOAD_B(j0, 3, bh, bl);
    WMMA_F16(c03, a0, bh);  WMMA_F16(c13, a1, bh);
    WMMA_F16(c03, a0, bl);  WMMA_F16(c13, a1, bl);

    d0 = e0; d1 = e1; d2 = e2; d3 = e3;
    f0 = g0; f1 = g1; f2 = g2; f3 = g3;
    j0 = jn;
  }
#undef LOAD_B
#undef LOAD_D

  // C/D layout: vgpr i, lane<16 -> M=i, N=lane ; lane>=16 -> M=8+i, N=lane-16
  const int mofs = (lane < 16) ? 0 : 8;
  const int ncol = lane & 15;
#pragma unroll
  for (int i = 0; i < 8; ++i) {
    int m0 = (mofs + i) * 64;          // M-tile 0 rows 0..15
    int m1 = (16 + mofs + i) * 64;     // M-tile 1 rows 16..31
    atomicAdd(&acc[m0 +      ncol], c00[i]);
    atomicAdd(&acc[m0 + 16 + ncol], c01[i]);
    atomicAdd(&acc[m0 + 32 + ncol], c02[i]);
    atomicAdd(&acc[m0 + 48 + ncol], c03[i]);
    atomicAdd(&acc[m1 +      ncol], c10[i]);
    atomicAdd(&acc[m1 + 16 + ncol], c11[i]);
    atomicAdd(&acc[m1 + 32 + ncol], c12[i]);
    atomicAdd(&acc[m1 + 48 + ncol], c13[i]);
  }
  __syncthreads();

  // write out, subtracting the diagonal (self) contribution exactly
  float* ao = agg + (size_t)i0 * 64;
  for (int k = tid; k < 2048; k += 256) {
    int m = k >> 6, c = k & 63;
    int r = i0 + m;
    float v = acc[k];
    if (dists[(size_t)r * n + r] < 1.5f) v -= msgs[(size_t)r * 64 + c];
    ao[k] = v;
  }
}

// ---------------------------------------------------------------------------
// Kernel 4: h2 = relu([h, agg] @ upd_W + b); logits = h2 @ out_W + b;
// gumbel via exact threefry2x32 replication of jax.random.uniform(key(42));
// softmax over 64.
// ---------------------------------------------------------------------------
__device__ __forceinline__ unsigned rotl32(unsigned x, int d) {
  return (x << d) | (x >> (32 - d));
}

__device__ float threefry_uniform(unsigned gidx, unsigned total) {
  const unsigned k0 = 0u, k1 = 42u;
  const unsigned k2 = 0x1BD11BDAu ^ k0 ^ k1;
  const unsigned half = total >> 1;
  const unsigned pair = (gidx < half) ? gidx : (gidx - half);
  unsigned x0 = pair + k0;
  unsigned x1 = (half + pair) + k1;
  const int r1[4] = {13, 15, 26, 6};
  const int r2[4] = {17, 29, 16, 24};
#pragma unroll
  for (int i = 0; i < 4; ++i) { x0 += x1; x1 = rotl32(x1, r1[i]); x1 ^= x0; }
  x0 += k1; x1 += k2 + 1u;
#pragma unroll
  for (int i = 0; i < 4; ++i) { x0 += x1; x1 = rotl32(x1, r2[i]); x1 ^= x0; }
  x0 += k2; x1 += k0 + 2u;
#pragma unroll
  for (int i = 0; i < 4; ++i) { x0 += x1; x1 = rotl32(x1, r1[i]); x1 ^= x0; }
  x0 += k0; x1 += k1 + 3u;
#pragma unroll
  for (int i = 0; i < 4; ++i) { x0 += x1; x1 = rotl32(x1, r2[i]); x1 ^= x0; }
  x0 += k1; x1 += k2 + 4u;
#pragma unroll
  for (int i = 0; i < 4; ++i) { x0 += x1; x1 = rotl32(x1, r1[i]); x1 ^= x0; }
  x0 += k2; x1 += k0 + 5u;
  unsigned bits = (gidx < half) ? x0 : x1;
  float f = __uint_as_float((bits >> 9) | 0x3f800000u) - 1.0f;
  const float lo = 1e-10f;
  float r = f * (1.0f - lo) + lo;
  return fmaxf(r, lo);
}

__global__ void __launch_bounds__(64)
head_kernel(const float* __restrict__ h, const float* __restrict__ agg,
            const float* __restrict__ updW, const float* __restrict__ updb,
            const float* __restrict__ outW, const float* __restrict__ outb,
            const float* __restrict__ tau_p, float* __restrict__ out, int n) {
  __shared__ float h2s[32];
  __shared__ float lg[64];
  __shared__ float es[64];
  const int row = blockIdx.x;
  const int t = threadIdx.x;
  const float tau = *tau_p;

  if (t < 32) {
    float s = updb[t];
    const float* hr = h + (size_t)row * 32;
#pragma unroll
    for (int k = 0; k < 32; ++k) s = fmaf(hr[k], updW[k * 32 + t], s);
    const float* ar = agg + (size_t)row * 64;
#pragma unroll
    for (int k = 0; k < 64; ++k) s = fmaf(ar[k], updW[(32 + k) * 32 + t], s);
    h2s[t] = fmaxf(s, 0.0f);
  }
  __syncthreads();

  float s = outb[t];
#pragma unroll
  for (int k = 0; k < 32; ++k) s = fmaf(h2s[k], outW[k * 64 + t], s);
  unsigned idx = (unsigned)row * 64u + (unsigned)t;
  float u = threefry_uniform(idx, (unsigned)n * 64u);
  float g = -logf(-logf(u));
  lg[t] = (s + g) / tau;
  __syncthreads();

  float mx = -INFINITY;
  for (int k = 0; k < 64; ++k) mx = fmaxf(mx, lg[k]);
  float e = expf(lg[t] - mx);
  es[t] = e;
  __syncthreads();
  float sum = 0.0f;
  for (int k = 0; k < 64; ++k) sum += es[k];
  out[(size_t)row * 64 + t] = e / sum;
}

// ---------------------------------------------------------------------------
extern "C" void kernel_launch(void* const* d_in, const int* in_sizes, int n_in,
                              void* d_out, int out_size, void* d_ws, size_t ws_size,
                              hipStream_t stream) {
  const float* z     = (const float*)d_in[0];
  const float* tau   = (const float*)d_in[1];
  const float* dists = (const float*)d_in[2];
  const float* encW  = (const float*)d_in[3];
  const float* encb  = (const float*)d_in[4];
  const float* msgW  = (const float*)d_in[5];
  const float* msgb  = (const float*)d_in[6];
  const float* updW  = (const float*)d_in[7];
  const float* updb  = (const float*)d_in[8];
  const float* outW  = (const float*)d_in[9];
  const float* outb  = (const float*)d_in[10];
  float* out = (float*)d_out;

  const int n = in_sizes[0] / 128;   // z is [N,128]

  char* ws = (char*)d_ws;
  float*    h      = (float*)ws;                            // n*32 f32
  _Float16* packHi = (_Float16*)(ws + (size_t)n * 128);     // n*64 f16
  _Float16* packLo = (_Float16*)(ws + (size_t)n * 256);     // n*64 f16
  float*    msgs   = (float*)(ws + (size_t)n * 384);        // n*64 f32
  float*    agg    = (float*)(ws + (size_t)n * 640);        // n*64 f32

  enc_kernel<<<(n * 32 + 255) / 256, 256, 0, stream>>>(z, encW, encb, h, n);
  msg_pack_kernel<<<(n * 64 + 255) / 256, 256, 0, stream>>>(h, msgW, msgb,
                                                            packHi, packLo, msgs, n);
  agg_wmma_kernel<<<n / 32, 256, 0, stream>>>(dists, packHi, packLo, msgs, agg, n);
  head_kernel<<<n, 64, 0, stream>>>(h, agg, updW, updb, outW, outb, tau, out, n);
}
